// DeepseekV3Experts_60894046323249
// MI455X (gfx1250) — compile-verified, matrix operand order checked by reading, not run on hardware
//
#include <hip/hip_runtime.h>

// ---------------- problem constants (from reference) ----------------
#define T_TOK 1024
#define HDIM  1024
#define IDIM  1408
#define NEXP  8
#define TOPK  4
#define TKROWS (T_TOK * TOPK)

// ---------------- GEMM tiling ----------------
#define M_TILE 64
#define N_TILE 32
#define KK     32
#define LDS_STRIDE 20   // uints per row: 16 data uints (32 bf16) + 4 pad (80B = 5*16B)

typedef __attribute__((ext_vector_type(16))) __bf16 v16bf;
typedef __attribute__((ext_vector_type(8)))  float  v8f;

union Frag {
    v16bf v;
    uint4 q[2];
};

// ---------------- CDNA5 async global->LDS path (guarded) ----------------
#if defined(__has_builtin)
#if __has_builtin(__builtin_amdgcn_global_load_async_to_lds_b128) && \
    __has_builtin(__builtin_amdgcn_s_wait_asynccnt)
#define USE_ASYNC_LDS 1
#endif
#if __has_builtin(__builtin_amdgcn_cvt_pk_bf16_f32)
#define USE_CVT_PK_BF16 1
#endif
#endif
#ifndef USE_ASYNC_LDS
#define USE_ASYNC_LDS 0
#endif
#ifndef USE_CVT_PK_BF16
#define USE_CVT_PK_BF16 0
#endif

#if USE_ASYNC_LDS
// Builtin signature (from hipcc diagnostic): params are pointers to
// gcc-vector 'int __attribute__((vector_size(16)))' in AS(1) / AS(3).
typedef int moe_v4i __attribute__((vector_size(16)));
typedef __attribute__((address_space(1))) moe_v4i* moe_gv4p;
typedef __attribute__((address_space(3))) moe_v4i* moe_lv4p;
#endif

__device__ __forceinline__ void copy16_g2l(const unsigned int* gsrc, unsigned int* ldst) {
#if USE_ASYNC_LDS
    __builtin_amdgcn_global_load_async_to_lds_b128(
        (moe_gv4p)gsrc, (moe_lv4p)ldst, 0, 0);
#else
    *(uint4*)ldst = *(const uint4*)gsrc;
#endif
}

__device__ __forceinline__ void async_wait_all() {
#if USE_ASYNC_LDS
    __builtin_amdgcn_s_wait_asynccnt(0);
#endif
}

// ---------------- helpers ----------------
__device__ __forceinline__ unsigned int pack_bf16(float a, float b) {
#if USE_CVT_PK_BF16
    auto v = __builtin_amdgcn_cvt_pk_bf16_f32(a, b);   // v_cvt_pk_bf16_f32 (RNE)
    unsigned int u;
    __builtin_memcpy(&u, &v, sizeof(u));
    return u;
#else
    // Native fptrunc f32->bf16 (RNE); backend selects the hw bf16 convert.
    union { __bf16 h[2]; unsigned int u; } cv;
    cv.h[0] = (__bf16)a;
    cv.h[1] = (__bf16)b;
    return cv.u;
#endif
}

__device__ __forceinline__ unsigned short f32_to_bf16(float a) {
    union { __bf16 h; unsigned short u; } cv;
    cv.h = (__bf16)a;
    return cv.u;
}

// ---------------- routing ----------------
__global__ void moe_zero_kernel(float* __restrict__ out, int n,
                                int* __restrict__ counts, int* __restrict__ cursor) {
    int i = blockIdx.x * blockDim.x + threadIdx.x;
    if (i < n) out[i] = 0.0f;
    if (i < NEXP) { counts[i] = 0; cursor[i] = 0; }
}

__global__ void moe_count_kernel(const int* __restrict__ sel, int* __restrict__ counts) {
    int i = blockIdx.x * blockDim.x + threadIdx.x;
    if (i < TKROWS) atomicAdd(&counts[sel[i]], 1);
}

__global__ void moe_scan_kernel(const int* __restrict__ counts, int* __restrict__ offsets) {
    if (threadIdx.x == 0 && blockIdx.x == 0) {
        int acc = 0;
        for (int e = 0; e < NEXP; ++e) { offsets[e] = acc; acc += counts[e]; }
        offsets[NEXP] = acc;
    }
}

__global__ void moe_scatter_kernel(const int* __restrict__ sel,
                                   const int* __restrict__ offsets,
                                   int* __restrict__ cursor,
                                   int* __restrict__ sorted) {
    int i = blockIdx.x * blockDim.x + threadIdx.x;
    if (i < TKROWS) {
        int e = sel[i];
        int pos = atomicAdd(&cursor[e], 1);
        sorted[offsets[e] + pos] = i;
    }
}

// ---------------- weight pre-pack: f32 [E,K,N] -> bf16 tile-contiguous ----------------
// Packed layout: tile (kc,nt) of 32x32 stored as 512 uints: [n][kp] at n*16+kp,
// uint = bf16 pair (k=kc*32+2*kp, k+1). Tile index: (e*NT + nt)*KC + kc.
// grid: (KC*NT, E), block 256.
__global__ __launch_bounds__(256)
void moe_pack_kernel(const float* __restrict__ src, unsigned int* __restrict__ dst,
                     int Kdim, int Ndim) {
    const int KC = Kdim >> 5;
    const int NT = Ndim >> 5;
    const int tile = blockIdx.x;
    const int e  = blockIdx.y;
    const int kc = tile % KC;
    const int nt = tile / KC;

    __shared__ unsigned int t_lds[512];

    const int tid = threadIdx.x;
    const int kp  = tid >> 4;          // 0..15 (k pair)
    const int c2  = (tid & 15) * 2;    // column 0..30 step 2

    const float* s = src + (size_t)e * Kdim * Ndim
                         + (size_t)(kc * 32 + 2 * kp) * Ndim + (nt * 32 + c2);
    float2 r0 = *(const float2*)s;
    float2 r1 = *(const float2*)(s + Ndim);
    t_lds[(c2 + 0) * 16 + kp] = pack_bf16(r0.x, r1.x);
    t_lds[(c2 + 1) * 16 + kp] = pack_bf16(r0.y, r1.y);
    __syncthreads();

    unsigned int* d = dst + ((size_t)(e * NT + nt) * KC + kc) * 512;
    *(uint2*)&d[tid * 2] = *(const uint2*)&t_lds[tid * 2];
}

// ---------------- fused gate/up GEMM + SwiGLU ----------------
// grid: (TKROWS/M_TILE, IDIM/N_TILE, NEXP), block: 256 (8 waves, 4x2 sub-tiles)
__global__ __launch_bounds__(256)
void moe_gateup_kernel(const float* __restrict__ hidden,
                       const unsigned int* __restrict__ pg,   // packed gate
                       const unsigned int* __restrict__ pu,   // packed up
                       const int* __restrict__ offsets,
                       const int* __restrict__ sorted,
                       unsigned short* __restrict__ h_buf) {
    const int e  = blockIdx.z;
    const int g0 = offsets[e];
    const int g1 = offsets[e + 1];
    const int m0 = blockIdx.x * M_TILE;
    if (g0 + m0 >= g1) return;
    const int nt = blockIdx.y;
    const int n0 = nt * N_TILE;
    const int KC = HDIM / KK;               // 32
    const int NT = IDIM / N_TILE;           // 44

    __shared__ unsigned int a_tile[2][M_TILE * LDS_STRIDE];
    __shared__ unsigned int bg_tile[2][N_TILE * LDS_STRIDE];
    __shared__ unsigned int bu_tile[2][N_TILE * LDS_STRIDE];

    const int tid  = threadIdx.x;
    const int lane = tid & 31;
    const int wave = tid >> 5;
    const int wm   = (wave & 3) * 16;
    const int wn   = (wave >> 2) * 16;

    // A staging role: row = tid>>2, 8 f32 elements per thread (convert to bf16)
    const int a_row = tid >> 2;
    const int a_cu  = (tid & 3) * 4;        // uint offset in LDS row
    const int a_el  = (tid & 3) * 8;        // f32 element offset
    const int rg = g0 + m0 + a_row;
    const bool a_valid = (rg < g1);
    int token = 0;
    if (a_valid) token = sorted[rg] >> 2;   // / TOPK
    const float* a_src = hidden + (size_t)token * HDIM + a_el;

    // B staging role: threads 0-127 -> gate tile, 128-255 -> up tile
    const int t2 = tid & 127;
    const int bn = t2 >> 2;                 // column 0..31
    const int bk = (t2 & 3) * 4;            // uint offset 0/4/8/12
    const unsigned int* b_pack =
        ((tid < 128) ? pg : pu) + ((size_t)(e * NT + nt) * KC) * 512 + bn * 16 + bk;
    unsigned int (*b_dst)[N_TILE * LDS_STRIDE] = (tid < 128) ? bg_tile : bu_tile;

    // zero the LDS rows of out-of-group A rows once (both buffers)
    if (!a_valid) {
        uint4 z = make_uint4(0u, 0u, 0u, 0u);
        *(uint4*)&a_tile[0][a_row * LDS_STRIDE + a_cu] = z;
        *(uint4*)&a_tile[1][a_row * LDS_STRIDE + a_cu] = z;
    }

    auto stage = [&](int kcIdx, int b) {
        if (a_valid) {
            const float* s = a_src + kcIdx * KK;
            float4 x0 = *(const float4*)s;
            float4 x1 = *(const float4*)(s + 4);
            uint4 pk;
            pk.x = pack_bf16(x0.x, x0.y);
            pk.y = pack_bf16(x0.z, x0.w);
            pk.z = pack_bf16(x1.x, x1.y);
            pk.w = pack_bf16(x1.z, x1.w);
            *(uint4*)&a_tile[b][a_row * LDS_STRIDE + a_cu] = pk;
        }
        copy16_g2l(b_pack + (size_t)kcIdx * 512, &b_dst[b][bn * LDS_STRIDE + bk]);
    };

    v8f acc_g = {};
    v8f acc_u = {};
    const int frag_r  = lane & 15;
    const int frag_kb = (lane >> 4) * 4;

    stage(0, 0);
    async_wait_all();
    __syncthreads();

    for (int kc = 0; kc < KC; ++kc) {
        const int cur = kc & 1;
        if (kc + 1 < KC) stage(kc + 1, cur ^ 1);

        Frag a, bg, bu;
        const unsigned int* ap  = &a_tile[cur][(wm + frag_r) * LDS_STRIDE + frag_kb];
        a.q[0]  = *(const uint4*)ap;
        a.q[1]  = *(const uint4*)(ap + 8);
        const unsigned int* bgp = &bg_tile[cur][(wn + frag_r) * LDS_STRIDE + frag_kb];
        bg.q[0] = *(const uint4*)bgp;
        bg.q[1] = *(const uint4*)(bgp + 8);
        const unsigned int* bup = &bu_tile[cur][(wn + frag_r) * LDS_STRIDE + frag_kb];
        bu.q[0] = *(const uint4*)bup;
        bu.q[1] = *(const uint4*)(bup + 8);

        acc_g = __builtin_amdgcn_wmma_f32_16x16x32_bf16(
            false, a.v, false, bg.v, (short)0, acc_g, false, false);
        acc_u = __builtin_amdgcn_wmma_f32_16x16x32_bf16(
            false, a.v, false, bu.v, (short)0, acc_u, false, false);

        async_wait_all();
        __syncthreads();
    }

    // ---- SwiGLU epilogue (v_rcp instead of IEEE divide), store bf16 h ----
    const int n_out  = n0 + wn + (lane & 15);
    const int m_base = wm + 8 * (lane >> 4);
    for (int r = 0; r < 8; ++r) {
        int rgl = g0 + m0 + m_base + r;
        if (rgl < g1) {
            float g = acc_g[r];
            float u = acc_u[r];
            float sig = __builtin_amdgcn_rcpf(1.0f + __expf(-g));
            h_buf[(size_t)rgl * IDIM + n_out] = f32_to_bf16(g * sig * u);
        }
    }
}

// ---------------- down GEMM + weighted scatter-combine ----------------
// grid: (TKROWS/M_TILE, HDIM/N_TILE, NEXP), block: 256
__global__ __launch_bounds__(256)
void moe_down_kernel(const unsigned short* __restrict__ h_buf,
                     const unsigned int* __restrict__ pd,     // packed down
                     const float* __restrict__ rweights,
                     const int* __restrict__ offsets,
                     const int* __restrict__ sorted,
                     float* __restrict__ out) {
    const int e  = blockIdx.z;
    const int g0 = offsets[e];
    const int g1 = offsets[e + 1];
    const int m0 = blockIdx.x * M_TILE;
    if (g0 + m0 >= g1) return;
    const int nt = blockIdx.y;
    const int n0 = nt * N_TILE;
    const int KC = IDIM / KK;               // 44
    const int NT = HDIM / N_TILE;           // 32

    __shared__ unsigned int a_tile[2][M_TILE * LDS_STRIDE];
    __shared__ unsigned int b_tile[2][N_TILE * LDS_STRIDE];

    const int tid  = threadIdx.x;
    const int lane = tid & 31;
    const int wave = tid >> 5;
    const int wm   = (wave & 3) * 16;
    const int wn   = (wave >> 2) * 16;

    // A staging (h already bf16): 16B per thread, async copy
    const int a_row = tid >> 2;
    const int a_cu  = (tid & 3) * 4;
    int rg = g0 + m0 + a_row;
    const bool a_valid = (rg < g1);
    if (rg >= TKROWS) rg = TKROWS - 1;
    const unsigned int* a_src =
        (const unsigned int*)h_buf + (size_t)rg * (IDIM / 2) + a_cu;

    // B staging: threads 0-127 copy the packed down tile
    const int bn = (tid & 127) >> 2;
    const int bk = ((tid & 127) & 3) * 4;
    const unsigned int* b_pack =
        pd + ((size_t)(e * NT + nt) * KC) * 512 + bn * 16 + bk;

    if (!a_valid) {
        uint4 z = make_uint4(0u, 0u, 0u, 0u);
        *(uint4*)&a_tile[0][a_row * LDS_STRIDE + a_cu] = z;
        *(uint4*)&a_tile[1][a_row * LDS_STRIDE + a_cu] = z;
    }

    auto stage = [&](int kcIdx, int b) {
        if (a_valid)
            copy16_g2l(a_src + kcIdx * (KK / 2), &a_tile[b][a_row * LDS_STRIDE + a_cu]);
        if (tid < 128)
            copy16_g2l(b_pack + (size_t)kcIdx * 512, &b_tile[b][bn * LDS_STRIDE + bk]);
    };

    v8f acc = {};
    const int frag_r  = lane & 15;
    const int frag_kb = (lane >> 4) * 4;

    stage(0, 0);
    async_wait_all();
    __syncthreads();

    for (int kc = 0; kc < KC; ++kc) {
        const int cur = kc & 1;
        if (kc + 1 < KC) stage(kc + 1, cur ^ 1);

        Frag a, b;
        const unsigned int* ap = &a_tile[cur][(wm + frag_r) * LDS_STRIDE + frag_kb];
        a.q[0] = *(const uint4*)ap;
        a.q[1] = *(const uint4*)(ap + 8);
        const unsigned int* bp = &b_tile[cur][(wn + frag_r) * LDS_STRIDE + frag_kb];
        b.q[0] = *(const uint4*)bp;
        b.q[1] = *(const uint4*)(bp + 8);

        acc = __builtin_amdgcn_wmma_f32_16x16x32_bf16(
            false, a.v, false, b.v, (short)0, acc, false, false);

        async_wait_all();
        __syncthreads();
    }

    const int n_out  = n0 + wn + (lane & 15);
    const int m_base = wm + 8 * (lane >> 4);
    for (int r = 0; r < 8; ++r) {
        int rgl = g0 + m0 + m_base + r;
        if (rgl < g1) {
            int p = sorted[rgl];
            float w = rweights[p];
            int t = p >> 2;  // / TOPK
            unsafeAtomicAdd(&out[(size_t)t * HDIM + n_out], w * acc[r]);
        }
    }
}

// ---------------- host launcher ----------------
extern "C" void kernel_launch(void* const* d_in, const int* in_sizes, int n_in,
                              void* d_out, int out_size, void* d_ws, size_t ws_size,
                              hipStream_t stream) {
    (void)in_sizes; (void)n_in; (void)out_size; (void)ws_size;

    const float* hidden   = (const float*)d_in[0];
    const float* rweights = (const float*)d_in[1];
    const float* gate_w   = (const float*)d_in[2];
    const float* up_w     = (const float*)d_in[3];
    const float* down_w   = (const float*)d_in[4];
    const int*   sel      = (const int*)d_in[5];
    float* out = (float*)d_out;

    // workspace layout (all 256B-aligned sections)
    char* ws = (char*)d_ws;
    int* counts  = (int*)ws;              // 8
    int* offsets = counts + 8;            // 9
    int* cursor  = counts + 20;           // 8
    int* sorted  = (int*)(ws + 256);      // TKROWS ints
    unsigned short* h_buf = (unsigned short*)(ws + 256 + TKROWS * sizeof(int));
    size_t off_pg = 256 + TKROWS * sizeof(int) + (size_t)TKROWS * IDIM * 2;
    const size_t psize = (size_t)NEXP * (HDIM / 2) * IDIM * 4;  // == NEXP*(IDIM/2)*HDIM*4
    unsigned int* pg = (unsigned int*)(ws + off_pg);
    unsigned int* pu = (unsigned int*)(ws + off_pg + psize);
    unsigned int* pd = (unsigned int*)(ws + off_pg + 2 * psize);

    const int nout = T_TOK * HDIM;
    moe_zero_kernel<<<(nout + 255) / 256, 256, 0, stream>>>(out, nout, counts, cursor);
    moe_count_kernel<<<(TKROWS + 255) / 256, 256, 0, stream>>>(sel, counts);
    moe_scan_kernel<<<1, 32, 0, stream>>>(counts, offsets);
    moe_scatter_kernel<<<(TKROWS + 255) / 256, 256, 0, stream>>>(sel, offsets, cursor, sorted);

    // pre-pack the three weight tensors into bf16 tile-contiguous form
    dim3 grid_pk_g((HDIM / 32) * (IDIM / 32), NEXP);   // (1408, 8)
    moe_pack_kernel<<<grid_pk_g, 256, 0, stream>>>(gate_w, pg, HDIM, IDIM);
    moe_pack_kernel<<<grid_pk_g, 256, 0, stream>>>(up_w,   pu, HDIM, IDIM);
    dim3 grid_pk_d((IDIM / 32) * (HDIM / 32), NEXP);   // (1408, 8)
    moe_pack_kernel<<<grid_pk_d, 256, 0, stream>>>(down_w, pd, IDIM, HDIM);

    dim3 grid_gu(TKROWS / M_TILE, IDIM / N_TILE, NEXP);   // (64, 44, 8)
    moe_gateup_kernel<<<grid_gu, 256, 0, stream>>>(hidden, pg, pu, offsets, sorted, h_buf);

    dim3 grid_dn(TKROWS / M_TILE, HDIM / N_TILE, NEXP);   // (64, 32, 8)
    moe_down_kernel<<<grid_dn, 256, 0, stream>>>(h_buf, pd, rweights, offsets, sorted, out);
}
